// IMCHAN_30889404793420
// MI455X (gfx1250) — compile-verified, matrix operand order checked by reading, not run on hardware
//
#include <hip/hip_runtime.h>
#include <hip/hip_bf16.h>

// ---------------------------------------------------------------------------
// HAN (2-branch, 2-metapath GAT + semantic attention) for MI455X / gfx1250.
// fp32 end-to-end; dense GEMMs via V_WMMA_F32_16X16X4_F32 (wave32 WMMA).
// Graph ops via ordered-key atomicMax (segment max) + float atomicAdd.
// ---------------------------------------------------------------------------

typedef __attribute__((ext_vector_type(2)))  float    v2f;
typedef __attribute__((ext_vector_type(8)))  float    v8f;
typedef __attribute__((ext_vector_type(16))) _Float16 v16h;

#define HAVE_WMMA_F32X4 __has_builtin(__builtin_amdgcn_wmma_f32_16x16x4_f32)

// ---------------- ordered-key float max (monotonic uint mapping) -----------
__device__ __forceinline__ unsigned fkey(float f) {
  unsigned b = __float_as_uint(f);
  return (b & 0x80000000u) ? ~b : (b | 0x80000000u);
}
__device__ __forceinline__ float fkey_inv(unsigned k) {
  return (k & 0x80000000u) ? __uint_as_float(k & 0x7fffffffu)
                           : __uint_as_float(~k);
}
#define KEY_NEG_INF 0x007FFFFFu  // fkey(-inf)

// ---------------------------------------------------------------------------
// GEMM: C[M x Nc] = A[M x K] * B.  TRANSB=0: B is [K x Nc] row-major.
// TRANSB=1: B is [Nc x K] row-major (i.e. multiply by B^T).
// Requires: K % 64 == 0, Nc % 16 == 0.  Block = 256 threads = 8 waves,
// block tile = 128 rows x 16 cols, each wave owns one 16x16 WMMA tile.
// ---------------------------------------------------------------------------
template <int TRANSB>
__global__ __launch_bounds__(256) void gemm_wmma_f32(
    const float* __restrict__ A, const float* __restrict__ B,
    float* __restrict__ C, int M, int Nc, int K) {
  __shared__ float As[128][65];  // +1 pad: conflict-free column reads
  __shared__ float Bs[64][17];

  const int tid = threadIdx.x;
  const int wave = tid >> 5;
  const int lane = tid & 31;
  const int rowBase = blockIdx.x * 128;
  const int colBase = blockIdx.y * 16;

  v8f acc = {};

  for (int k0 = 0; k0 < K; k0 += 64) {
    // ---- stage A chunk (128 x 64) : 2048 float4, 8 per thread ----
    for (int i = tid; i < 2048; i += 256) {
      int r  = i >> 4;
      int c4 = (i & 15) << 2;
      int gr = rowBase + r;
      float4 v = make_float4(0.f, 0.f, 0.f, 0.f);
      if (gr < M) v = *(const float4*)(A + (size_t)gr * K + k0 + c4);
      As[r][c4 + 0] = v.x; As[r][c4 + 1] = v.y;
      As[r][c4 + 2] = v.z; As[r][c4 + 3] = v.w;
    }
    // ---- stage B chunk (64 x 16) ----
    for (int i = tid; i < 1024; i += 256) {
      int kk = i >> 4, c = i & 15;
      float v;
      if (TRANSB) v = B[(size_t)(colBase + c) * K + k0 + kk];
      else        v = B[(size_t)(k0 + kk) * Nc + colBase + c];
      Bs[kk][c] = v;
    }
    // prefetch next A chunk into cache hierarchy (global_prefetch_b8)
    if (k0 + 64 < K) {
      int pr = rowBase + (tid & 127);
      if (pr < M) __builtin_prefetch(A + (size_t)pr * K + k0 + 64, 0, 1);
    }
    __syncthreads();

    const int rw = wave * 16;
#if HAVE_WMMA_F32X4
    // fp32 WMMA: A 16x4 (lanes 0-15 hold K0,K1; 16-31 hold K2,K3),
    // B 4x16 symmetric.  16 MACs per 64-chunk.
#pragma unroll
    for (int kk = 0; kk < 64; kk += 4) {
      v2f a, b;
      if (lane < 16) {
        a.x = As[rw + lane][kk + 0]; a.y = As[rw + lane][kk + 1];
        b.x = Bs[kk + 0][lane];      b.y = Bs[kk + 1][lane];
      } else {
        a.x = As[rw + lane - 16][kk + 2]; a.y = As[rw + lane - 16][kk + 3];
        b.x = Bs[kk + 2][lane - 16];      b.y = Bs[kk + 3][lane - 16];
      }
      acc = __builtin_amdgcn_wmma_f32_16x16x4_f32(
          false, a, false, b, (short)0, acc, false, false);
    }
#else
    // fallback: fp16 fragments into confirmed wmma_f32_16x16x32_f16
#pragma unroll
    for (int kk = 0; kk < 64; kk += 32) {
      v16h a, b;
      const int r_  = rw + (lane & 15);
      const int ko  = (lane < 16) ? 0 : 8;
      const int c_  = lane & 15;
#pragma unroll
      for (int j = 0; j < 8; ++j) {
        a[j]     = (_Float16)As[r_][kk + ko + j];
        a[8 + j] = (_Float16)As[r_][kk + 16 + ko + j];
        b[j]     = (_Float16)Bs[kk + ko + j][c_];
        b[8 + j] = (_Float16)Bs[kk + 16 + ko + j][c_];
      }
      acc = __builtin_amdgcn_wmma_f32_16x16x32_f16(
          false, a, false, b, (short)0, acc, false, false);
    }
#endif
    __syncthreads();
  }

  // ---- store: C/D layout — lanes 0-15: M=r, lanes 16-31: M=r+8 ----
  const int col  = colBase + (lane & 15);
  const int rofs = (lane < 16) ? 0 : 8;
#pragma unroll
  for (int r = 0; r < 8; ++r) {
    int row = rowBase + wave * 16 + rofs + r;
    if (row < M) C[(size_t)row * Nc + col] = acc[r];
  }
}

// ---------------------------------------------------------------------------
// el[n,h] = sum_k feat[n,h,k]*al[h,k];  er likewise.  One wave per node.
// ---------------------------------------------------------------------------
__global__ __launch_bounds__(256) void elr_kernel(
    const float* __restrict__ feat, const float* __restrict__ al,
    const float* __restrict__ ar, float* __restrict__ el,
    float* __restrict__ er, int N) {
  int n = (blockIdx.x * blockDim.x + threadIdx.x) >> 5;
  int lane = threadIdx.x & 31;
  if (n >= N) return;
  const float* row = feat + (size_t)n * 512;
#pragma unroll
  for (int h = 0; h < 8; ++h) {
    float x1 = row[h * 64 + lane], x2 = row[h * 64 + 32 + lane];
    float sl = x1 * al[h * 64 + lane] + x2 * al[h * 64 + 32 + lane];
    float sr = x1 * ar[h * 64 + lane] + x2 * ar[h * 64 + 32 + lane];
#pragma unroll
    for (int off = 16; off; off >>= 1) {
      sl += __shfl_xor(sl, off, 32);
      sr += __shfl_xor(sr, off, 32);
    }
    if (lane == 0) { el[n * 8 + h] = sl; er[n * 8 + h] = sr; }
  }
}

// ---------------------------------------------------------------------------
__global__ __launch_bounds__(256) void init_gat(
    float* __restrict__ rst, float* __restrict__ ssum,
    unsigned* __restrict__ keybuf, int N) {
  int i = blockIdx.x * blockDim.x + threadIdx.x;
  if (i < N * 512) rst[i] = 0.f;
  if (i < N * 8) { ssum[i] = 0.f; keybuf[i] = KEY_NEG_INF; }
}

__global__ __launch_bounds__(256) void edge_max(
    const int* __restrict__ src, const int* __restrict__ dst,
    const float* __restrict__ el, const float* __restrict__ er,
    unsigned* __restrict__ keybuf, int E) {
  int i = blockIdx.x * blockDim.x + threadIdx.x;
  if (i >= E * 8) return;
  int e = i >> 3, h = i & 7;
  int s = src[e], d = dst[e];
  float v = el[s * 8 + h] + er[d * 8 + h];
  v = v > 0.f ? v : 0.2f * v;  // leaky_relu(0.2)
  atomicMax(&keybuf[d * 8 + h], fkey(v));
}

__global__ __launch_bounds__(256) void finalize_max(
    const unsigned* __restrict__ keybuf, float* __restrict__ mbuf, int N) {
  int i = blockIdx.x * blockDim.x + threadIdx.x;
  if (i >= N * 8) return;
  float f = fkey_inv(keybuf[i]);
  mbuf[i] = (f - f == 0.0f) ? f : 0.0f;  // where(isfinite(m), m, 0)
}

__global__ __launch_bounds__(256) void edge_exp(
    const int* __restrict__ src, const int* __restrict__ dst,
    const float* __restrict__ el, const float* __restrict__ er,
    const float* __restrict__ mbuf, float* __restrict__ exbuf,
    float* __restrict__ ssum, int E) {
  int i = blockIdx.x * blockDim.x + threadIdx.x;
  if (i >= E * 8) return;
  int e = i >> 3, h = i & 7;
  int s = src[e], d = dst[e];
  float v = el[s * 8 + h] + er[d * 8 + h];
  v = v > 0.f ? v : 0.2f * v;
  float ex = __expf(v - mbuf[d * 8 + h]);
  exbuf[i] = ex;
  atomicAdd(&ssum[d * 8 + h], ex);
}

// One wave per edge: lane owns 16 consecutive feature elems (h = lane>>2
// is lane-constant), gathers feat[src] with b128 loads, scatters atomicAdd.
__global__ __launch_bounds__(256) void edge_aggregate(
    const int* __restrict__ src, const int* __restrict__ dst,
    const float* __restrict__ exbuf, const float* __restrict__ ssum,
    const float* __restrict__ feat, float* __restrict__ rst, int E) {
  int e = (blockIdx.x * blockDim.x + threadIdx.x) >> 5;
  int lane = threadIdx.x & 31;
  if (e >= E) return;
  int s = src[e], d = dst[e];
  int h = lane >> 2;
  float alpha = exbuf[(size_t)e * 8 + h] / ssum[d * 8 + h];
  const float4* fs = (const float4*)(feat + (size_t)s * 512 + lane * 16);
  float* rd = rst + (size_t)d * 512 + lane * 16;
#pragma unroll
  for (int i = 0; i < 4; ++i) {
    float4 v = fs[i];
    atomicAdd(rd + 4 * i + 0, v.x * alpha);
    atomicAdd(rd + 4 * i + 1, v.y * alpha);
    atomicAdd(rd + 4 * i + 2, v.z * alpha);
    atomicAdd(rd + 4 * i + 3, v.w * alpha);
  }
}

__global__ __launch_bounds__(256) void bias_elu(
    float* __restrict__ z, const float* __restrict__ bias, int total) {
  int i = blockIdx.x * blockDim.x + threadIdx.x;
  if (i >= total) return;
  float v = z[i] + bias[i & 511];
  z[i] = v > 0.f ? v : (__expf(v) - 1.f);
}

// ---------------------------------------------------------------------------
// Semantic attention: per node (one block of 128):
//   w_m = sum_j tanh(z_m . W1[:,j] + b1[j]) * W2[j];  beta = softmax(w);
//   fused = beta0*z0 + beta1*z1
// ---------------------------------------------------------------------------
__global__ __launch_bounds__(128) void semantic_fuse(
    const float* __restrict__ z0, const float* __restrict__ z1,
    const float* __restrict__ W1, const float* __restrict__ b1,
    const float* __restrict__ W2, float* __restrict__ fused, int N) {
  __shared__ float zs[2][512];
  __shared__ float red[128];
  __shared__ float beta[2];
  int n = blockIdx.x, t = threadIdx.x;
  for (int k = t; k < 512; k += 128) {
    zs[0][k] = z0[(size_t)n * 512 + k];
    zs[1][k] = z1[(size_t)n * 512 + k];
  }
  __syncthreads();
  float w[2];
#pragma unroll
  for (int m = 0; m < 2; ++m) {
    float acc = b1[t];
    for (int k = 0; k < 512; ++k) acc += zs[m][k] * W1[(size_t)k * 128 + t];
    red[t] = tanhf(acc) * W2[t];
    __syncthreads();
    for (int off = 64; off; off >>= 1) {
      if (t < off) red[t] += red[t + off];
      __syncthreads();
    }
    w[m] = red[0];
    __syncthreads();
  }
  if (t == 0) {
    float mx = fmaxf(w[0], w[1]);
    float e0 = __expf(w[0] - mx), e1 = __expf(w[1] - mx);
    beta[0] = e0 / (e0 + e1);
    beta[1] = e1 / (e0 + e1);
  }
  __syncthreads();
  float b0 = beta[0], b1v = beta[1];
  for (int k = t; k < 512; k += 128)
    fused[(size_t)n * 512 + k] = b0 * zs[0][k] + b1v * zs[1][k];
}

// ---------------------------------------------------------------------------
extern "C" void kernel_launch(void* const* d_in, const int* in_sizes, int n_in,
                              void* d_out, int out_size, void* d_ws,
                              size_t ws_size, hipStream_t stream) {
  const int F = 256, W = 256, D = 512;
  const int N = in_sizes[0] / F;     // 10000
  const int E = in_sizes[20] / 2;    // 160000

  // ---- workspace carve (all L2-resident; ~79 MB) ----
  float* ws = (float*)d_ws;
  size_t o = 0;
  float* wh   = ws + o; o += (size_t)N * 256;
  float* feat = ws + o; o += (size_t)N * 512;
  float* z0   = ws + o; o += (size_t)N * 512;
  float* z1   = ws + o; o += (size_t)N * 512;
  float* el   = ws + o; o += (size_t)N * 8;
  float* er   = ws + o; o += (size_t)N * 8;
  unsigned* keybuf = (unsigned*)(ws + o); o += (size_t)N * 8;
  float* mbuf = ws + o; o += (size_t)N * 8;
  float* ssum = ws + o; o += (size_t)N * 8;
  float* exbf = ws + o; o += (size_t)E * 8;
  float* fused = feat;  // feat is dead by semantic-attention time

  const float* X[2] = {(const float*)d_in[0], (const float*)d_in[1]};
  const int pbase[2] = {2, 11};
  const int* eptr[2][2] = {{(const int*)d_in[20], (const int*)d_in[21]},
                           {(const int*)d_in[22], (const int*)d_in[23]}};
  float* outH[2] = {(float*)d_out, (float*)d_out + (size_t)N * 64};
  float* prod = (float*)d_out + (size_t)2 * N * 64;

  const int gM = (N + 127) / 128;              // 79
  const dim3 blk256(256), blk128(128);
  const int edgeBlk = (E * 8 + 255) / 256;     // per-(edge,head) kernels
  const int waveEdgeBlk = (E + 7) / 8;         // wave-per-edge kernel
  const int nodeWaveBlk = (N + 7) / 8;         // wave-per-node kernel
  const int nd512Blk = (N * 512 + 255) / 256;
  const int nhBlk = (N * 8 + 255) / 256;

  for (int br = 0; br < 2; ++br) {
    const int b = pbase[br];
    const float* w_h  = (const float*)d_in[b + 0];
    const float* gfc  = (const float*)d_in[b + 1];  // [2,256,512]
    const float* gal  = (const float*)d_in[b + 2];  // [2,8,64]
    const float* gar  = (const float*)d_in[b + 3];
    const float* gbia = (const float*)d_in[b + 4];  // [2,512]
    const float* sW1  = (const float*)d_in[b + 5];  // [512,128]
    const float* sb1  = (const float*)d_in[b + 6];  // [128]
    const float* sW2  = (const float*)d_in[b + 7];  // [128]
    const float* pred = (const float*)d_in[b + 8];  // [512,64]

    // wh = x @ w_h : [N,256]x[256,256]
    gemm_wmma_f32<0><<<dim3(gM, W / 16), blk256, 0, stream>>>(
        X[br], w_h, wh, N, W, F);

    float* zbuf[2] = {z0, z1};
    for (int m = 0; m < 2; ++m) {
      const int* src = eptr[br][m];
      const int* dst = eptr[br][m] + E;
      // feat = wh @ fc[m] : [N,256]x[256,512]
      gemm_wmma_f32<0><<<dim3(gM, D / 16), blk256, 0, stream>>>(
          wh, gfc + (size_t)m * W * D, feat, N, D, W);
      elr_kernel<<<nodeWaveBlk, blk256, 0, stream>>>(
          feat, gal + (size_t)m * 512, gar + (size_t)m * 512, el, er, N);
      init_gat<<<nd512Blk, blk256, 0, stream>>>(zbuf[m], ssum, keybuf, N);
      edge_max<<<edgeBlk, blk256, 0, stream>>>(src, dst, el, er, keybuf, E);
      finalize_max<<<nhBlk, blk256, 0, stream>>>(keybuf, mbuf, N);
      edge_exp<<<edgeBlk, blk256, 0, stream>>>(
          src, dst, el, er, mbuf, exbf, ssum, E);
      edge_aggregate<<<waveEdgeBlk, blk256, 0, stream>>>(
          src, dst, exbf, ssum, feat, zbuf[m], E);
      bias_elu<<<nd512Blk, blk256, 0, stream>>>(
          zbuf[m], gbia + (size_t)m * 512, N * 512);
    }

    semantic_fuse<<<N, blk128, 0, stream>>>(z0, z1, sW1, sb1, sW2, fused, N);
    // out_b = fused @ pred : [N,512]x[512,64]
    gemm_wmma_f32<0><<<dim3(gM, 64 / 16), blk256, 0, stream>>>(
        fused, pred, outH[br], N, 64, D);
  }

  // prod = h1 @ h2^T : [N,64] x [N,64]^T -> [N,N]
  gemm_wmma_f32<1><<<dim3(gM, N / 16), blk256, 0, stream>>>(
      outH[0], outH[1], prod, N, N, 64);
}
